// GATGT_50002009260140
// MI455X (gfx1250) — matrix operand order repaired and checked by reading, not compiled, not run on hardware
//
#include <hip/hip_runtime.h>
#include <stdint.h>
#include <stddef.h>

// ---------------- problem constants ----------------
#define NNODES   6720
#define NGRAPHS  16
#define NPGR     420
#define NEDGE    53760
#define NHEADS   4
#define INDIM    2048
#define CH1      1024
#define HC1      4096          // NHEADS*CH1
#define CH2      64
#define OUTDIM   18
#define NEXT     (NEDGE + NNODES)   // edges + self loops
#define SLOPE    0.2f

typedef __attribute__((ext_vector_type(16))) _Float16 v16h;
typedef __attribute__((ext_vector_type(8)))  _Float16 v8h;
typedef __attribute__((ext_vector_type(8)))  float    v8f;

// ---------------- small helpers ----------------
__device__ __forceinline__ unsigned fenc(float f) {
  unsigned b = __float_as_uint(f);
  return (b & 0x80000000u) ? ~b : (b | 0x80000000u);
}
__device__ __forceinline__ float fdec(unsigned e) {
  return (e & 0x80000000u) ? __uint_as_float(e ^ 0x80000000u)
                           : __uint_as_float(~e);
}
__device__ __forceinline__ float wave_sum(float v) {
  #pragma unroll
  for (int off = 16; off; off >>= 1) v += __shfl_down(v, off, 32);
  return v;
}

// ---------------- generic fill ----------------
__global__ void fill_u32(uint32_t* __restrict__ p, uint32_t v, size_t n) {
  size_t i = (size_t)blockIdx.x * blockDim.x + threadIdx.x;
  size_t s = (size_t)gridDim.x * blockDim.x;
  for (; i < n; i += s) p[i] = v;
}

// ---------------- f32 -> f16 convert ----------------
__global__ void cvt_f2h(const float* __restrict__ s, _Float16* __restrict__ d, size_t n) {
  size_t i = (size_t)blockIdx.x * blockDim.x + threadIdx.x;
  size_t st = (size_t)gridDim.x * blockDim.x;
  for (; i < n; i += st) d[i] = (_Float16)s[i];
}

// ---------------- transpose + convert: W[K,N] f32 -> Wt[N,K] f16 ----------------
__global__ __launch_bounds__(256)
void transpose_cvt(const float* __restrict__ S, _Float16* __restrict__ D, int K, int Nn) {
  __shared__ float t[32][33];
  int kb = blockIdx.x * 32, nb = blockIdx.y * 32;
  int tx = threadIdx.x, ty = threadIdx.y;      // 32 x 8
  #pragma unroll
  for (int i = 0; i < 32; i += 8) {
    int k = kb + ty + i, n = nb + tx;
    t[ty + i][tx] = (k < K && n < Nn) ? S[(size_t)k * Nn + n] : 0.f;
  }
  __syncthreads();
  #pragma unroll
  for (int i = 0; i < 32; i += 8) {
    int n = nb + ty + i, k = kb + tx;
    if (n < Nn && k < K) D[(size_t)n * K + k] = (_Float16)t[tx][ty + i];
  }
}

// ---------------- WMMA fragment loaders (16-bit, 16x16x32) ----------------
// A frag (16x32): lane holds row (lane&15); k-chunks [kb,kb+8) and [kb+16,kb+24), kb = 8*(lane>>4)
__device__ __forceinline__ v16h ldfragA(const _Float16* __restrict__ p) {
  v8h lo = *(const v8h*)p;
  v8h hh = *(const v8h*)(p + 16);
  v16h r;
  #pragma unroll
  for (int i = 0; i < 8; i++) { r[i] = lo[i]; r[i + 8] = hh[i]; }
  return r;
}
// B frag (32x16) from Bt[N,K]: lane holds col (lane&15); k-chunks [kb,kb+8) and [kb+16,kb+24), kb = 8*(lane>>4)
__device__ __forceinline__ v16h ldfragB(const _Float16* __restrict__ p) {
  v8h lo = *(const v8h*)p;
  v8h hh = *(const v8h*)(p + 16);
  v16h r;
  #pragma unroll
  for (int i = 0; i < 8; i++) { r[i] = lo[i]; r[i + 8] = hh[i]; }
  return r;
}

// ---------------- WMMA GEMM: C[M,N] = A[M,K](f16) * Bt[N,K](f16)^T + bias ----------------
// block 256 = 8 waves in 4(M) x 2(N); wave tile 32x64 (2x4 WMMA frags); block tile 128x128.
// 8 WMMAs per 12 B128 loads per K=32 step.
template <bool OUT_HALF>
__global__ __launch_bounds__(256)
void wmma_gemm(const _Float16* __restrict__ A, const _Float16* __restrict__ Bt,
               const float* __restrict__ bias, void* __restrict__ Cout,
               int M, int N, int K) {
  const int lane = threadIdx.x & 31;
  const int wave = threadIdx.x >> 5;
  const int wm = wave & 3, wn = wave >> 2;
  const int row0 = blockIdx.x * 128 + wm * 32;
  const int col0 = blockIdx.y * 128 + wn * 64;
  const int lr = lane & 15;
  const int hi = lane >> 4;

  const int ra0 = row0 + lr, ra1 = row0 + 16 + lr;
  const _Float16* a0p = A + (size_t)(ra0 < M ? ra0 : 0) * K + hi * 8;
  const _Float16* a1p = A + (size_t)(ra1 < M ? ra1 : 0) * K + hi * 8;
  const _Float16* bp[4];
  #pragma unroll
  for (int t = 0; t < 4; t++) {
    int nb = col0 + t * 16 + lr;
    bp[t] = Bt + (size_t)(nb < N ? nb : 0) * K + hi * 8;
  }

  v8f acc[2][4] = {};

  for (int k0 = 0; k0 < K; k0 += 32) {
    __builtin_prefetch(a0p + k0 + 128, 0, 0);
    __builtin_prefetch(bp[0] + k0 + 128, 0, 0);
    v16h af0 = ldfragA(a0p + k0);
    v16h af1 = ldfragA(a1p + k0);
    v16h bf[4];
    #pragma unroll
    for (int t = 0; t < 4; t++) bf[t] = ldfragB(bp[t] + k0);
    #pragma unroll
    for (int t = 0; t < 4; t++) {
      acc[0][t] = __builtin_amdgcn_wmma_f32_16x16x32_f16(false, af0, false, bf[t],
                                                         (short)0, acc[0][t], false, false);
      acc[1][t] = __builtin_amdgcn_wmma_f32_16x16x32_f16(false, af1, false, bf[t],
                                                         (short)0, acc[1][t], false, false);
    }
  }

  // C/D layout: vgpr j, lanes 0-15 -> (m=j, n=lane), lanes 16-31 -> (m=j+8, n=lane-16)
  #pragma unroll
  for (int tm = 0; tm < 2; tm++) {
    #pragma unroll
    for (int tn = 0; tn < 4; tn++) {
      int n = col0 + tn * 16 + lr;
      if (n < N) {
        float bv = bias ? bias[n] : 0.f;
        int mbase = row0 + tm * 16 + hi * 8;
        v8f a = acc[tm][tn];
        #pragma unroll
        for (int j = 0; j < 8; j++) {
          int m = mbase + j;
          if (m < M) {
            float val = a[j] + bv;
            if (OUT_HALF) ((_Float16*)Cout)[(size_t)m * N + n] = (_Float16)val;
            else          ((float*)Cout)[(size_t)m * N + n] = val;
          }
        }
      }
    }
  }
}

// ---------------- edge dot scores (TransformerConv): one wave per (e,h) ----------------
__global__ __launch_bounds__(256)
void edge_scores_tc(const _Float16* __restrict__ q, const _Float16* __restrict__ k,
                    const int* __restrict__ src, const int* __restrict__ dst,
                    float* __restrict__ scores, int E, int H, int C, float scale) {
  int wid = (int)(((size_t)blockIdx.x * blockDim.x + threadIdx.x) >> 5);
  int lane = threadIdx.x & 31;
  if (wid >= E * H) return;
  int e = wid / H, h = wid - e * H;
  int s = src[e], d = dst[e];
  const _Float16* qp = q + (size_t)d * H * C + h * C;
  const _Float16* kp = k + (size_t)s * H * C + h * C;
  float acc = 0.f;
  for (int c = lane * 8; c < C; c += 256) {
    v8h a = *(const v8h*)(qp + c);
    v8h b = *(const v8h*)(kp + c);
    #pragma unroll
    for (int j = 0; j < 8; j++) acc += (float)a[j] * (float)b[j];
  }
  acc = wave_sum(acc);
  if (lane == 0) scores[wid] = acc * scale;
}

// ---------------- GAT node scores: one wave per (n,h) ----------------
__global__ __launch_bounds__(256)
void gat_node_scores(const _Float16* __restrict__ hmat, const float* __restrict__ a_src,
                     const float* __restrict__ a_dst, float* __restrict__ es,
                     float* __restrict__ ed, int Nn, int H, int C) {
  int wid = (int)(((size_t)blockIdx.x * blockDim.x + threadIdx.x) >> 5);
  int lane = threadIdx.x & 31;
  if (wid >= Nn * H) return;
  int n = wid / H, h = wid - n * H;
  const _Float16* hp = hmat + (size_t)n * H * C + h * C;
  const float* as = a_src + h * C;
  const float* ad = a_dst + h * C;
  float s1 = 0.f, s2 = 0.f;
  for (int c = lane; c < C; c += 32) {
    float hv = (float)hp[c];
    s1 += hv * as[c];
    s2 += hv * ad[c];
  }
  s1 = wave_sum(s1);
  s2 = wave_sum(s2);
  if (lane == 0) { es[wid] = s1; ed[wid] = s2; }
}

// ---------------- build extended (self-loop) edge arrays ----------------
__global__ void build_ext(const int* __restrict__ src, const int* __restrict__ dst,
                          int* __restrict__ xs, int* __restrict__ xd, int E, int Nn) {
  int i = blockIdx.x * blockDim.x + threadIdx.x;
  if (i < E) { xs[i] = src[i]; xd[i] = dst[i]; }
  else if (i < E + Nn) { xs[i] = i - E; xd[i] = i - E; }
}

// ---------------- GAT edge scores (leaky relu) ----------------
__global__ void gat_edge_scores(const float* __restrict__ es, const float* __restrict__ ed,
                                const int* __restrict__ xs, const int* __restrict__ xd,
                                float* __restrict__ scores, int E, int H) {
  int i = blockIdx.x * blockDim.x + threadIdx.x;
  if (i >= E * H) return;
  int e = i / H, h = i - e * H;
  float v = es[xs[e] * H + h] + ed[xd[e] * H + h];
  scores[i] = v > 0.f ? v : SLOPE * v;
}

// ---------------- segment softmax primitives ----------------
__global__ void seg_max(const float* __restrict__ scores, const int* __restrict__ dst,
                        unsigned* __restrict__ mx, int E, int H) {
  int i = blockIdx.x * blockDim.x + threadIdx.x;
  if (i >= E * H) return;
  int e = i / H, h = i - e * H;
  atomicMax(&mx[dst[e] * H + h], fenc(scores[i]));
}
__global__ void seg_expsum(float* __restrict__ scores, const int* __restrict__ dst,
                           const unsigned* __restrict__ mx, float* __restrict__ sum,
                           int E, int H) {
  int i = blockIdx.x * blockDim.x + threadIdx.x;
  if (i >= E * H) return;
  int e = i / H, h = i - e * H;
  int key = dst[e] * H + h;
  float ex = __expf(scores[i] - fdec(mx[key]));
  scores[i] = ex;
  atomicAdd(&sum[key], ex);
}
__global__ void seg_norm(float* __restrict__ scores, const int* __restrict__ dst,
                         const float* __restrict__ sum, int E, int H) {
  int i = blockIdx.x * blockDim.x + threadIdx.x;
  if (i >= E * H) return;
  int e = i / H, h = i - e * H;
  scores[i] = scores[i] / (sum[dst[e] * H + h] + 1e-16f);
}

// ---------------- weighted scatter aggregation: one wave per (e,h) ----------------
__global__ __launch_bounds__(256)
void edge_aggregate(const _Float16* __restrict__ vals, const float* __restrict__ alpha,
                    const int* __restrict__ src, const int* __restrict__ dst,
                    float* __restrict__ out, int E, int H, int C) {
  int wid = (int)(((size_t)blockIdx.x * blockDim.x + threadIdx.x) >> 5);
  int lane = threadIdx.x & 31;
  if (wid >= E * H) return;
  int e = wid / H, h = wid - e * H;
  float a = alpha[wid];
  const _Float16* vp = vals + (size_t)src[e] * H * C + h * C;
  float* op = out + (size_t)dst[e] * H * C + h * C;
  for (int c = lane; c < C; c += 32) atomicAdd(op + c, a * (float)vp[c]);
}

// ---------------- elementwise epilogues ----------------
__global__ void add_relu_h(const float* __restrict__ a, const _Float16* __restrict__ b,
                           _Float16* __restrict__ o, size_t n) {
  size_t i = (size_t)blockIdx.x * blockDim.x + threadIdx.x;
  size_t s = (size_t)gridDim.x * blockDim.x;
  for (; i < n; i += s) o[i] = (_Float16)fmaxf(a[i] + (float)b[i], 0.f);
}
__global__ void addbias_relu_h(const float* __restrict__ a, const float* __restrict__ bias,
                               int nb, _Float16* __restrict__ o, size_t n) {
  size_t i = (size_t)blockIdx.x * blockDim.x + threadIdx.x;
  size_t s = (size_t)gridDim.x * blockDim.x;
  for (; i < n; i += s) o[i] = (_Float16)fmaxf(a[i] + bias[i % nb], 0.f);
}
__global__ void add_relu_f(const float* __restrict__ a, const float* __restrict__ b,
                           float* __restrict__ o, size_t n) {
  size_t i = (size_t)blockIdx.x * blockDim.x + threadIdx.x;
  size_t s = (size_t)gridDim.x * blockDim.x;
  for (; i < n; i += s) o[i] = fmaxf(a[i] + b[i], 0.f);
}
__global__ void addbias_relu_f(const float* __restrict__ a, const float* __restrict__ bias,
                               int nb, float* __restrict__ o, size_t n) {
  size_t i = (size_t)blockIdx.x * blockDim.x + threadIdx.x;
  size_t s = (size_t)gridDim.x * blockDim.x;
  for (; i < n; i += s) o[i] = fmaxf(a[i] + bias[i % nb], 0.f);
}
__global__ void add_f(const float* __restrict__ a, const float* __restrict__ b,
                      float* __restrict__ o, size_t n) {
  size_t i = (size_t)blockIdx.x * blockDim.x + threadIdx.x;
  size_t s = (size_t)gridDim.x * blockDim.x;
  for (; i < n; i += s) o[i] = a[i] + b[i];
}

// ---------------- tiny FC head: block per (out_col, graph) ----------------
__global__ __launch_bounds__(256)
void small_fc(const float* __restrict__ X, const float* __restrict__ W,
              const float* __restrict__ b, float* __restrict__ Y,
              int K, int Nout, int do_relu) {
  int j = blockIdx.x, g = blockIdx.y;
  __shared__ float red[256];
  float acc = 0.f;
  for (int kk = threadIdx.x; kk < K; kk += 256)
    acc += X[(size_t)g * K + kk] * W[(size_t)kk * Nout + j];
  red[threadIdx.x] = acc;
  __syncthreads();
  for (int s = 128; s; s >>= 1) {
    if ((int)threadIdx.x < s) red[threadIdx.x] += red[threadIdx.x + s];
    __syncthreads();
  }
  if (threadIdx.x == 0) {
    float v = red[0] + b[j];
    Y[(size_t)g * Nout + j] = do_relu ? fmaxf(v, 0.f) : v;
  }
}

// ================= host orchestration =================
static inline int cdiv(long long a, long long b) { return (int)((a + b - 1) / b); }

extern "C" void kernel_launch(void* const* d_in, const int* in_sizes, int n_in,
                              void* d_out, int out_size, void* d_ws, size_t ws_size,
                              hipStream_t stream) {
  // ---- inputs (setup_inputs dict order) ----
  const float* x      = (const float*)d_in[0];
  const int*   ei     = (const int*)d_in[1];
  const float* t1_w[4] = { (const float*)d_in[2], (const float*)d_in[3],
                           (const float*)d_in[4], (const float*)d_in[5] };   // wq wk wv ws
  const float* t1_b[4] = { (const float*)d_in[6], (const float*)d_in[7],
                           (const float*)d_in[8], (const float*)d_in[9] };   // bq bk bv bs
  const float* t2_w[4] = { (const float*)d_in[10], (const float*)d_in[11],
                           (const float*)d_in[12], (const float*)d_in[13] };
  const float* t2_b[4] = { (const float*)d_in[14], (const float*)d_in[15],
                           (const float*)d_in[16], (const float*)d_in[17] };
  const float* g1_w  = (const float*)d_in[18];
  const float* g1_as = (const float*)d_in[19];
  const float* g1_ad = (const float*)d_in[20];
  const float* g1_b  = (const float*)d_in[21];
  const float* g2_w  = (const float*)d_in[22];
  const float* g2_as = (const float*)d_in[23];
  const float* g2_ad = (const float*)d_in[24];
  const float* g2_b  = (const float*)d_in[25];
  const float* fc1_w = (const float*)d_in[26]; const float* fc1_b = (const float*)d_in[27];
  const float* fc2_w = (const float*)d_in[28]; const float* fc2_b = (const float*)d_in[29];
  const float* fc3_w = (const float*)d_in[30]; const float* fc3_b = (const float*)d_in[31];
  const float* fc4_w = (const float*)d_in[32]; const float* fc4_b = (const float*)d_in[33];
  const int* src = ei;
  const int* dst = ei + NEDGE;

  // ---- workspace carve-up ----
  char* p = (char*)d_ws;
  auto alloc = [&](size_t bytes) -> void* {
    void* r = (void*)p;
    p += (bytes + 255) & ~(size_t)255;
    return r;
  };
  _Float16* xh   = (_Float16*)alloc((size_t)NNODES * INDIM * 2);
  _Float16* wt   = (_Float16*)alloc((size_t)HC1 * INDIM * 2);      // transposed weight buffer (reused)
  _Float16* qh   = (_Float16*)alloc((size_t)NNODES * HC1 * 2);
  _Float16* kh   = (_Float16*)alloc((size_t)NNODES * HC1 * 2);
  _Float16* vh   = (_Float16*)alloc((size_t)NNODES * HC1 * 2);
  _Float16* sh   = (_Float16*)alloc((size_t)NNODES * HC1 * 2);
  float*    acc  = (float*)alloc((size_t)NNODES * HC1 * 4);        // reused accumulator
  _Float16* h1h  = (_Float16*)alloc((size_t)NNODES * HC1 * 2);
  float*    scores = (float*)alloc((size_t)NEXT * NHEADS * 4);
  unsigned* segmx  = (unsigned*)alloc((size_t)NNODES * NHEADS * 4);
  float*    segsum = (float*)alloc((size_t)NNODES * NHEADS * 4);
  float*    esbuf  = (float*)alloc((size_t)NNODES * NHEADS * 4);
  float*    edbuf  = (float*)alloc((size_t)NNODES * NHEADS * 4);
  int*      xsrc   = (int*)alloc((size_t)NEXT * 4);
  int*      xdst   = (int*)alloc((size_t)NEXT * 4);
  _Float16* q2h  = (_Float16*)alloc((size_t)NNODES * CH2 * 2);
  _Float16* k2h  = (_Float16*)alloc((size_t)NNODES * CH2 * 2);
  _Float16* v2h  = (_Float16*)alloc((size_t)NNODES * CH2 * 2);
  float*    s2f  = (float*)alloc((size_t)NNODES * CH2 * 4);
  _Float16* hz2h = (_Float16*)alloc((size_t)NNODES * CH2 * 2);
  float*    acc2 = (float*)alloc((size_t)NNODES * CH2 * 4);
  float*    h2f  = (float*)alloc((size_t)NNODES * CH2 * 4);
  float*    z2f  = (float*)alloc((size_t)NNODES * CH2 * 4);
  float*    sfeat= (float*)alloc((size_t)NNODES * CH2 * 4);
  float*    a1   = (float*)alloc((size_t)NGRAPHS * 256 * 4);
  float*    a2   = (float*)alloc((size_t)NGRAPHS * 128 * 4);
  float*    a3   = (float*)alloc((size_t)NGRAPHS * 64 * 4);
  _Float16* hgh = qh;   // reuse after transformer1 complete
  _Float16* z1h = kh;   // reuse after transformer1 complete

  const dim3 B256(256), Btr(32, 8);
  const dim3 Gbig(cdiv(NNODES, 128), HC1 / 128);
  const dim3 Gsmall(cdiv(NNODES, 128), 1);
  const dim3 GtrBig(INDIM / 32, HC1 / 32);     // 2048 x 4096
  const dim3 GtrT2(HC1 / 32, CH2 / 32);        // 4096 x 64

  // ---- phase 0: convert x ----
  size_t nx = (size_t)NNODES * INDIM;
  cvt_f2h<<<cdiv(nx, 256), B256, 0, stream>>>(x, xh, nx);

  // ---- phase 1: transformer conv 1 GEMMs (q,k,v,skip) ----
  _Float16* t1out[4] = { qh, kh, vh, sh };
  for (int i = 0; i < 4; i++) {
    transpose_cvt<<<GtrBig, Btr, 0, stream>>>(t1_w[i], wt, INDIM, HC1);
    wmma_gemm<true><<<Gbig, B256, 0, stream>>>(xh, wt, t1_b[i], t1out[i],
                                               NNODES, HC1, INDIM);
  }

  // ---- phase 2: transformer conv 1 attention ----
  {
    int EH = NEDGE * NHEADS, NH = NNODES * NHEADS;
    edge_scores_tc<<<cdiv((size_t)EH * 32, 256), B256, 0, stream>>>(
        qh, kh, src, dst, scores, NEDGE, NHEADS, CH1, 1.0f / 32.0f);
    fill_u32<<<cdiv(NH, 256), B256, 0, stream>>>((uint32_t*)segmx, 0u, NH);
    fill_u32<<<cdiv(NH, 256), B256, 0, stream>>>((uint32_t*)segsum, 0u, NH);
    seg_max<<<cdiv(EH, 256), B256, 0, stream>>>(scores, dst, segmx, NEDGE, NHEADS);
    seg_expsum<<<cdiv(EH, 256), B256, 0, stream>>>(scores, dst, segmx, segsum, NEDGE, NHEADS);
    seg_norm<<<cdiv(EH, 256), B256, 0, stream>>>(scores, dst, segsum, NEDGE, NHEADS);
    size_t na = (size_t)NNODES * HC1;
    fill_u32<<<cdiv(na, 256), B256, 0, stream>>>((uint32_t*)acc, 0u, na);
    edge_aggregate<<<cdiv((size_t)EH * 32, 256), B256, 0, stream>>>(
        vh, scores, src, dst, acc, NEDGE, NHEADS, CH1);
    add_relu_h<<<cdiv(na, 256), B256, 0, stream>>>(acc, sh, h1h, na);
  }

  // ---- phase 3: GAT conv 1 (reuses qh as hgh, kh as z1h) ----
  {
    transpose_cvt<<<GtrBig, Btr, 0, stream>>>(g1_w, wt, INDIM, HC1);
    wmma_gemm<true><<<Gbig, B256, 0, stream>>>(xh, wt, nullptr, hgh, NNODES, HC1, INDIM);
    gat_node_scores<<<cdiv((size_t)NNODES * NHEADS * 32, 256), B256, 0, stream>>>(
        hgh, g1_as, g1_ad, esbuf, edbuf, NNODES, NHEADS, CH1);
    build_ext<<<cdiv(NEXT, 256), B256, 0, stream>>>(src, dst, xsrc, xdst, NEDGE, NNODES);
    int EH = NEXT * NHEADS, NH = NNODES * NHEADS;
    gat_edge_scores<<<cdiv(EH, 256), B256, 0, stream>>>(esbuf, edbuf, xsrc, xdst,
                                                        scores, NEXT, NHEADS);
    fill_u32<<<cdiv(NH, 256), B256, 0, stream>>>((uint32_t*)segmx, 0u, NH);
    fill_u32<<<cdiv(NH, 256), B256, 0, stream>>>((uint32_t*)segsum, 0u, NH);
    seg_max<<<cdiv(EH, 256), B256, 0, stream>>>(scores, xdst, segmx, NEXT, NHEADS);
    seg_expsum<<<cdiv(EH, 256), B256, 0, stream>>>(scores, xdst, segmx, segsum, NEXT, NHEADS);
    seg_norm<<<cdiv(EH, 256), B256, 0, stream>>>(scores, xdst, segsum, NEXT, NHEADS);
    size_t na = (size_t)NNODES * HC1;
    fill_u32<<<cdiv(na, 256), B256, 0, stream>>>((uint32_t*)acc, 0u, na);
    edge_aggregate<<<cdiv((size_t)EH * 32, 256), B256, 0, stream>>>(
        hgh, scores, xsrc, xdst, acc, NEXT, NHEADS, CH1);
    addbias_relu_h<<<cdiv(na, 256), B256, 0, stream>>>(acc, g1_b, HC1, z1h, na);
  }

  // ---- phase 4: transformer conv 2 (h1 -> 64, 1 head) ----
  {
    transpose_cvt<<<GtrT2, Btr, 0, stream>>>(t2_w[0], wt, HC1, CH2);
    wmma_gemm<true><<<Gsmall, B256, 0, stream>>>(h1h, wt, t2_b[0], q2h, NNODES, CH2, HC1);
    transpose_cvt<<<GtrT2, Btr, 0, stream>>>(t2_w[1], wt, HC1, CH2);
    wmma_gemm<true><<<Gsmall, B256, 0, stream>>>(h1h, wt, t2_b[1], k2h, NNODES, CH2, HC1);
    transpose_cvt<<<GtrT2, Btr, 0, stream>>>(t2_w[2], wt, HC1, CH2);
    wmma_gemm<true><<<Gsmall, B256, 0, stream>>>(h1h, wt, t2_b[2], v2h, NNODES, CH2, HC1);
    transpose_cvt<<<GtrT2, Btr, 0, stream>>>(t2_w[3], wt, HC1, CH2);
    wmma_gemm<false><<<Gsmall, B256, 0, stream>>>(h1h, wt, t2_b[3], s2f, NNODES, CH2, HC1);

    edge_scores_tc<<<cdiv((size_t)NEDGE * 32, 256), B256, 0, stream>>>(
        q2h, k2h, src, dst, scores, NEDGE, 1, CH2, 1.0f / 8.0f);
    fill_u32<<<cdiv(NNODES, 256), B256, 0, stream>>>((uint32_t*)segmx, 0u, NNODES);
    fill_u32<<<cdiv(NNODES, 256), B256, 0, stream>>>((uint32_t*)segsum, 0u, NNODES);
    seg_max<<<cdiv(NEDGE, 256), B256, 0, stream>>>(scores, dst, segmx, NEDGE, 1);
    seg_expsum<<<cdiv(NEDGE, 256), B256, 0, stream>>>(scores, dst, segmx, segsum, NEDGE, 1);
    seg_norm<<<cdiv(NEDGE, 256), B256, 0, stream>>>(scores, dst, segsum, NEDGE, 1);
    size_t na = (size_t)NNODES * CH2;
    fill_u32<<<cdiv(na, 256), B256, 0, stream>>>((uint32_t*)acc2, 0u, na);
    edge_aggregate<<<cdiv((size_t)NEDGE * 32, 256), B256, 0, stream>>>(
        v2h, scores, src, dst, acc2, NEDGE, 1, CH2);
    add_relu_f<<<cdiv(na, 256), B256, 0, stream>>>(acc2, s2f, h2f, na);
  }

  // ---- phase 5: GAT conv 2 (z1 -> 64, 1 head, mean==identity) ----
  {
    transpose_cvt<<<GtrT2, Btr, 0, stream>>>(g2_w, wt, HC1, CH2);
    wmma_gemm<true><<<Gsmall, B256, 0, stream>>>(z1h, wt, nullptr, hz2h, NNODES, CH2, HC1);
    gat_node_scores<<<cdiv((size_t)NNODES * 32, 256), B256, 0, stream>>>(
        hz2h, g2_as, g2_ad, esbuf, edbuf, NNODES, 1, CH2);
    gat_edge_scores<<<cdiv(NEXT, 256), B256, 0, stream>>>(esbuf, edbuf, xsrc, xdst,
                                                          scores, NEXT, 1);
    fill_u32<<<cdiv(NNODES, 256), B256, 0, stream>>>((uint32_t*)segmx, 0u, NNODES);
    fill_u32<<<cdiv(NNODES, 256), B256, 0, stream>>>((uint32_t*)segsum, 0u, NNODES);
    seg_max<<<cdiv(NEXT, 256), B256, 0, stream>>>(scores, xdst, segmx, NEXT, 1);
    seg_expsum<<<cdiv(NEXT, 256), B256, 0, stream>>>(scores, xdst, segmx, segsum, NEXT, 1);
    seg_norm<<<cdiv(NEXT, 256), B256, 0, stream>>>(scores, xdst, segsum, NEXT, 1);
    size_t na = (size_t)NNODES * CH2;
    fill_u32<<<cdiv(na, 256), B256, 0, stream>>>((uint32_t*)acc2, 0u, na);
    edge_aggregate<<<cdiv((size_t)NEXT * 32, 256), B256, 0, stream>>>(
        hz2h, scores, xsrc, xdst, acc2, NEXT, 1, CH2);
    addbias_relu_f<<<cdiv(na, 256), B256, 0, stream>>>(acc2, g2_b, CH2, z2f, na);
  }

  // ---- phase 6: combine + MLP head ----
  {
    size_t na = (size_t)NNODES * CH2;
    add_f<<<cdiv(na, 256), B256, 0, stream>>>(h2f, z2f, sfeat, na);
    small_fc<<<dim3(256, NGRAPHS), B256, 0, stream>>>(sfeat, fc1_w, fc1_b, a1,
                                                      NPGR * CH2, 256, 1);
    small_fc<<<dim3(128, NGRAPHS), B256, 0, stream>>>(a1, fc2_w, fc2_b, a2, 256, 128, 1);
    small_fc<<<dim3(64, NGRAPHS), B256, 0, stream>>>(a2, fc3_w, fc3_b, a3, 128, 64, 1);
    small_fc<<<dim3(OUTDIM, NGRAPHS), B256, 0, stream>>>(a3, fc4_w, fc4_b, (float*)d_out,
                                                         64, OUTDIM, 0);
  }
  (void)in_sizes; (void)n_in; (void)out_size; (void)ws_size;
}
